// TransformerMoELayer_2920577761558
// MI455X (gfx1250) — matrix operand order repaired and checked by reading, not compiled, not run on hardware
//
#include <hip/hip_runtime.h>

typedef __bf16 bf16;
typedef __attribute__((ext_vector_type(16))) __bf16 v16bf;
typedef __attribute__((ext_vector_type(8)))  float  v8f;

#define DEVINL __device__ __forceinline__

constexpr int kB  = 4;
constexpr int kS  = 2048;
constexpr int kH  = 1024;
constexpr int kNH = 16;
constexpr int kD  = 64;
constexpr int kE  = 8;
constexpr int kF  = 4096;
constexpr int kCAP = 2048;           // ceil(8192/8 * 2 * 1.0)
constexpr int kBS = kB * kS;         // 8192 tokens
constexpr int kH3 = 3 * kH;          // 3072

DEVINL bf16 f2bf(float f) {
  unsigned u = __float_as_uint(f);
  u = u + 0x7FFFu + ((u >> 16) & 1u);                 // round-to-nearest-even
  unsigned short s = (unsigned short)(u >> 16);
  return __builtin_bit_cast(bf16, s);
}

DEVINL unsigned short bf_bits(float f) {
  unsigned u = __float_as_uint(f);
  u = u + 0x7FFFu + ((u >> 16) & 1u);
  return (unsigned short)(u >> 16);
}

DEVINL v8f v8f_zero() {
  v8f z;
#pragma unroll
  for (int i = 0; i < 8; i++) z[i] = 0.0f;
  return z;
}

union FragBF { v16bf v; uint4 u[2]; };
union U4BF   { uint4 u; bf16 h[8]; };

// ---------------------------------------------------------------------------
// fp32 -> bf16 conversion, 8 elements / thread-step (float4 x2 -> uint4)
// n8 = element count / 8 (all our sizes are multiples of 8)
// ---------------------------------------------------------------------------
__global__ void cvt_kernel(const float* __restrict__ in, bf16* __restrict__ out, long long n8) {
  long long i = (long long)blockIdx.x * blockDim.x + threadIdx.x;
  const long long stride = (long long)gridDim.x * blockDim.x;
  for (; i < n8; i += stride) {
    const float4* s = (const float4*)in + 2 * i;
    const float4 f0 = s[0];
    const float4 f1 = s[1];
    U4BF o;
    o.h[0] = f2bf(f0.x); o.h[1] = f2bf(f0.y); o.h[2] = f2bf(f0.z); o.h[3] = f2bf(f0.w);
    o.h[4] = f2bf(f1.x); o.h[5] = f2bf(f1.y); o.h[6] = f2bf(f1.z); o.h[7] = f2bf(f1.w);
    ((uint4*)out)[i] = o.u;
  }
}

// ---------------------------------------------------------------------------
// Generic tiled WMMA GEMM.  C[M,N] = act(A[M,K] * B + bias), batched over z.
//   BT=true : B stored [N][K] (C = A * B^T, contiguous K)
//   BT=false: B stored [K][N]
//   ACT : 0 = none, 1 = exact GELU
//   OUTBF : write bf16 (Cb) else fp32 (Cf)
// Block = 256 threads (8 waves), tile 128x128, K-step 32, double-buffered LDS:
// each iteration fetches tile k+1 into registers, computes WMMAs from buffer
// `cur`, stages the fetched tile into buffer `cur^1`, one barrier/iteration.
// ---------------------------------------------------------------------------
template <bool BT, int ACT, bool OUTBF>
__launch_bounds__(256)
__global__ void gemm_wmma_kernel(const bf16* __restrict__ A, const bf16* __restrict__ Bm,
                                 const float* __restrict__ bias,
                                 float* __restrict__ Cf, bf16* __restrict__ Cb,
                                 int N, int K,
                                 long long sA, long long sB, long long sBias, long long sC) {
  __shared__ bf16 As[2][128][48];   // stride 48: 16B-aligned frags, spread banks
  __shared__ bf16 Bs[2][128][48];   // always stored [n][k]

  const int bz = blockIdx.z;
  A  += (long long)bz * sA;
  Bm += (long long)bz * sB;
  const float* biasp = bias ? (bias + (long long)bz * sBias) : nullptr;
  const long long cOff = (long long)bz * sC;

  const int bn = blockIdx.x * 128;
  const int bm = blockIdx.y * 128;
  const int tid  = threadIdx.x;
  const int lane = tid & 31;
  const int wave = tid >> 5;
  const int wm = wave >> 1;          // 0..3
  const int wn = wave & 1;           // 0..1
  const int ln16 = lane & 15;
  const int kh   = lane >> 4;        // lane half

  // staging thread mapping
  const int rowL = tid >> 1;                    // A / B(BT): 128 rows
  const int kcL  = (tid & 1) << 4;
  const int kkL  = tid >> 3;                    // B(!BT): 32 k-rows
  const int ncL  = (tid & 7) << 4;              // x 128 n

  const bf16* aBase  = A  + (long long)(bm + rowL) * K + kcL;
  const bf16* bBaseT = Bm + (long long)(bn + rowL) * K + kcL;   // BT
  const bf16* bBaseN = Bm + (long long)kkL * N + (bn + ncL);    // !BT

  uint4 ar0, ar1;
  U4BF  br0, br1;

  auto fetch = [&](int k0) {
    const uint4* sa = (const uint4*)(aBase + k0);
    ar0 = sa[0]; ar1 = sa[1];
    if (BT) {
      const uint4* sb = (const uint4*)(bBaseT + k0);
      br0.u = sb[0]; br1.u = sb[1];
    } else {
      const uint4* sb = (const uint4*)(bBaseN + (long long)k0 * N);
      br0.u = sb[0]; br1.u = sb[1];
    }
  };
  auto stage = [&](int buf) {
    *(uint4*)&As[buf][rowL][kcL]     = ar0;
    *(uint4*)&As[buf][rowL][kcL + 8] = ar1;
    if (BT) {
      *(uint4*)&Bs[buf][rowL][kcL]     = br0.u;
      *(uint4*)&Bs[buf][rowL][kcL + 8] = br1.u;
    } else {
#pragma unroll
      for (int i = 0; i < 8; i++) {
        Bs[buf][ncL + i][kkL]     = br0.h[i];
        Bs[buf][ncL + 8 + i][kkL] = br1.h[i];
      }
    }
  };

  v8f acc[2][4];
#pragma unroll
  for (int i = 0; i < 2; i++)
#pragma unroll
    for (int j = 0; j < 4; j++) acc[i][j] = v8f_zero();

  const int nk = K >> 5;
  fetch(0);
  stage(0);
  __syncthreads();

  for (int ki = 0; ki < nk; ki++) {
    const int cur = ki & 1;
    if (ki + 1 < nk) fetch((ki + 1) << 5);           // next tile -> registers
    if (ki + 2 < nk) {                               // +2 tile -> GL2 prefetch
      const int k2 = (ki + 2) << 5;
      __builtin_prefetch(aBase + k2, 0, 3);
      if (BT) __builtin_prefetch(bBaseT + k2, 0, 3);
      else    __builtin_prefetch(bBaseN + (long long)k2 * N, 0, 3);
    }

    FragBF a[2], bfr[4];
#pragma unroll
    for (int mi = 0; mi < 2; mi++) {
      const bf16* p = &As[cur][wm * 32 + mi * 16 + ln16][kh * 8];
      a[mi].u[0] = *(const uint4*)p;          // K = kh*8 + 0..7
      a[mi].u[1] = *(const uint4*)(p + 16);   // K = 16 + kh*8 + 0..7
    }
#pragma unroll
    for (int ni = 0; ni < 4; ni++) {
      const bf16* p = &Bs[cur][wn * 64 + ni * 16 + ln16][kh * 8];
      bfr[ni].u[0] = *(const uint4*)p;        // K = kh*8 + 0..7
      bfr[ni].u[1] = *(const uint4*)(p + 16);
    }
#pragma unroll
    for (int mi = 0; mi < 2; mi++)
#pragma unroll
      for (int ni = 0; ni < 4; ni++)
        acc[mi][ni] = __builtin_amdgcn_wmma_f32_16x16x32_bf16(
            false, a[mi].v, false, bfr[ni].v, (short)0, acc[mi][ni], false, false);

    if (ki + 1 < nk) stage(cur ^ 1);
    __syncthreads();
  }

  // epilogue: C element r -> row = kh*8 + r, col = ln16
#pragma unroll
  for (int mi = 0; mi < 2; mi++) {
#pragma unroll
    for (int ni = 0; ni < 4; ni++) {
      const int col = bn + wn * 64 + ni * 16 + ln16;
      const float bv = biasp ? biasp[col] : 0.0f;
#pragma unroll
      for (int r = 0; r < 8; r++) {
        const int row = bm + wm * 32 + mi * 16 + kh * 8 + r;
        float v = acc[mi][ni][r] + bv;
        if (ACT == 1) v = 0.5f * v * (1.0f + erff(v * 0.70710678118654752f));
        const long long idx = cOff + (long long)row * N + col;
        if (OUTBF) Cb[idx] = f2bf(v);
        else       Cf[idx] = v;
      }
    }
  }
}

// ---------------------------------------------------------------------------
// Flash attention over qkv buffer [B,S,3H] (bf16), head dim 64.
// Grid: (S/128, NH, B); block 256 = 8 waves, each wave owns 16 q-rows.
// Online softmax; K/V chunks of 32 double-buffered in LDS; V stored with
// key->slot permutation slot = 2*(key%16) + key/16 so P-probabilities of one
// lane are adjacent in K-order (packed b32 stores); P re-laid out via LDS.
// ---------------------------------------------------------------------------
__launch_bounds__(256)
__global__ void attn_flash_kernel(const bf16* __restrict__ qkv, bf16* __restrict__ out) {
  __shared__ bf16 Ksh[2][32][64];     // [key][d]
  __shared__ bf16 Vsh[2][64][48];     // [d][slot] transposed+permuted
  __shared__ bf16 Psh[8][16][32];     // per-wave P tile (K-order = slot order)

  const int qb = blockIdx.x * 128;
  const int h  = blockIdx.y;
  const int b  = blockIdx.z;
  const int tid  = threadIdx.x;
  const int lane = tid & 31;
  const int wave = tid >> 5;
  const int ln16 = lane & 15;
  const int kh   = lane >> 4;

  // Q fragments for this wave's 16 rows (d = 64 -> two K=32 A-frags)
  const long long rowQ = (long long)b * kS + qb + wave * 16 + ln16;
  const bf16* qptr = qkv + rowQ * kH3 + h * kD;
  FragBF qa[2];
#pragma unroll
  for (int dh = 0; dh < 2; dh++) {
    const bf16* p = qptr + dh * 32 + kh * 16;
    qa[dh].u[0] = *(const uint4*)p;
    qa[dh].u[1] = *(const uint4*)(p + 8);
  }

  // K/V staging mapping: thread -> (key row rS, d col cS)
  const int rS = tid >> 3;                          // key 0..31
  const int cS = (tid & 7) << 3;                    // d 0..56
  const int slotS = ((rS & 15) << 1) | (rS >> 4);   // key -> slot permutation
  const bf16* kBase = qkv + ((long long)b * kS + rS) * kH3 + kH     + h * kD + cS;
  const bf16* vBase = qkv + ((long long)b * kS + rS) * kH3 + 2 * kH + h * kD + cS;

  uint4 kreg; U4BF vreg;
  auto fetchKV = [&](int kc) {
    const long long o = (long long)kc * kH3;
    kreg   = *(const uint4*)(kBase + o);
    vreg.u = *(const uint4*)(vBase + o);
  };
  auto stageKV = [&](int buf) {
    *(uint4*)&Ksh[buf][rS][cS] = kreg;
#pragma unroll
    for (int i = 0; i < 8; i++) Vsh[buf][cS + i][slotS] = vreg.h[i];
  };

  float mrow[8], lrow[8];
  v8f oacc[4];
#pragma unroll
  for (int r = 0; r < 8; r++) { mrow[r] = -3.0e38f; lrow[r] = 0.0f; }
#pragma unroll
  for (int t = 0; t < 4; t++) oacc[t] = v8f_zero();

  const int nchunks = kS / 32;
  fetchKV(0);
  stageKV(0);
  __syncthreads();

  for (int ci = 0; ci < nchunks; ci++) {
    const int cur = ci & 1;
    const int kc  = ci * 32;
    if (ci + 1 < nchunks) fetchKV(kc + 32);
    if (ci + 2 < nchunks) {
      const long long o2 = (long long)(kc + 64) * kH3;
      __builtin_prefetch(kBase + o2, 0, 3);
      __builtin_prefetch(vBase + o2, 0, 3);
    }

    // scores: 16 q-rows x 32 keys = two 16x16 tiles, each over d=64 (2 wmma)
    v8f s0 = v8f_zero(), s1 = v8f_zero();
#pragma unroll
    for (int dh = 0; dh < 2; dh++) {
      FragBF kb;
      const bf16* p = &Ksh[cur][ln16][dh * 32 + kh * 16];
      kb.u[0] = *(const uint4*)p; kb.u[1] = *(const uint4*)(p + 8);
      s0 = __builtin_amdgcn_wmma_f32_16x16x32_bf16(false, qa[dh].v, false, kb.v, (short)0, s0, false, false);
    }
#pragma unroll
    for (int dh = 0; dh < 2; dh++) {
      FragBF kb;
      const bf16* p = &Ksh[cur][16 + ln16][dh * 32 + kh * 16];
      kb.u[0] = *(const uint4*)p; kb.u[1] = *(const uint4*)(p + 8);
      s1 = __builtin_amdgcn_wmma_f32_16x16x32_bf16(false, qa[dh].v, false, kb.v, (short)0, s1, false, false);
    }

    // online softmax per row (row = kh*8 + r); 16-lane groups share a row set
#pragma unroll
    for (int r = 0; r < 8; r++) {
      const float a0 = s0[r] * 0.125f;     // 1/sqrt(64)
      const float a1 = s1[r] * 0.125f;
      float cm = fmaxf(a0, a1);
#pragma unroll
      for (int off = 1; off < 16; off <<= 1) cm = fmaxf(cm, __shfl_xor(cm, off, 32));
      const float nm = fmaxf(mrow[r], cm);
      const float alpha = __expf(mrow[r] - nm);
      mrow[r] = nm;
      const float p0 = __expf(a0 - nm);    // key ln16      -> slot 2*ln16
      const float p1 = __expf(a1 - nm);    // key 16+ln16   -> slot 2*ln16+1
      float rs = p0 + p1;
#pragma unroll
      for (int off = 1; off < 16; off <<= 1) rs += __shfl_xor(rs, off, 32);
      lrow[r] = lrow[r] * alpha + rs;
#pragma unroll
      for (int t = 0; t < 4; t++) oacc[t][r] *= alpha;
      const unsigned ppack = (unsigned)bf_bits(p0) | ((unsigned)bf_bits(p1) << 16);
      *(unsigned*)&Psh[wave][kh * 8 + r][2 * ln16] = ppack;
    }

    asm volatile("s_wait_dscnt 0" ::: "memory");  // P stores visible before re-read

    FragBF pa;
    {
      const bf16* p = &Psh[wave][ln16][kh * 16];
      pa.u[0] = *(const uint4*)p;
      pa.u[1] = *(const uint4*)(p + 8);
    }
#pragma unroll
    for (int t = 0; t < 4; t++) {
      FragBF vb;
      const bf16* p = &Vsh[cur][t * 16 + ln16][kh * 16];
      vb.u[0] = *(const uint4*)p; vb.u[1] = *(const uint4*)(p + 8);
      oacc[t] = __builtin_amdgcn_wmma_f32_16x16x32_bf16(false, pa.v, false, vb.v, (short)0, oacc[t], false, false);
    }

    if (ci + 1 < nchunks) stageKV(cur ^ 1);
    __syncthreads();
  }

  // normalize and store bf16 into attn buffer [B,S,H]
#pragma unroll
  for (int t = 0; t < 4; t++) {
    const int col = h * kD + t * 16 + ln16;
#pragma unroll
    for (int r = 0; r < 8; r++) {
      const long long row = (long long)b * kS + qb + wave * 16 + kh * 8 + r;
      out[row * kH + col] = f2bf(oacc[t][r] / lrow[r]);
    }
  }
}

// ---------------------------------------------------------------------------
// out = LayerNorm(x + y) * w + b ;  optional bf16 copy
// one block (256 threads) per row of 1024
// ---------------------------------------------------------------------------
__launch_bounds__(256)
__global__ void add_ln_kernel(const float* __restrict__ x, const float* __restrict__ y,
                              const float* __restrict__ w, const float* __restrict__ bia,
                              float* __restrict__ outf, bf16* __restrict__ outb) {
  const long long row = blockIdx.x;
  const float* xr = x + row * kH;
  const float* yr = y + row * kH;
  float v[4]; float s = 0.f, sq = 0.f;
#pragma unroll
  for (int i = 0; i < 4; i++) {
    const int c = threadIdx.x + i * 256;
    v[i] = xr[c] + yr[c];
    s += v[i]; sq += v[i] * v[i];
  }
#pragma unroll
  for (int off = 1; off < 32; off <<= 1) { s += __shfl_xor(s, off, 32); sq += __shfl_xor(sq, off, 32); }
  __shared__ float ss[8], sq2[8];
  const int wave = threadIdx.x >> 5;
  if ((threadIdx.x & 31) == 0) { ss[wave] = s; sq2[wave] = sq; }
  __syncthreads();
  float ts = 0.f, tq = 0.f;
#pragma unroll
  for (int i = 0; i < 8; i++) { ts += ss[i]; tq += sq2[i]; }
  const float mu  = ts * (1.0f / 1024.0f);
  const float var = tq * (1.0f / 1024.0f) - mu * mu;
  const float rs  = rsqrtf(var + 1e-5f);
#pragma unroll
  for (int i = 0; i < 4; i++) {
    const int c = threadIdx.x + i * 256;
    const float o = (v[i] - mu) * rs * w[c] + bia[c];
    outf[row * kH + c] = o;
    if (outb) outb[row * kH + c] = f2bf(o);
  }
}

// ---------------------------------------------------------------------------
// top-2 gating: logits, softmax, argmax1 / argmax2 (first-occurrence ties)
// block = 256 threads -> 32 tokens x 8 experts
// ---------------------------------------------------------------------------
__launch_bounds__(256)
__global__ void gate_kernel(const float* __restrict__ x1, const float* __restrict__ gw,
                            int* __restrict__ idx1, int* __restrict__ idx2,
                            float* __restrict__ g1, float* __restrict__ g2) {
  __shared__ float lg[32][9];
  const int t  = threadIdx.x;
  const int tl = t >> 3;
  const int e  = t & 7;
  const int tok = blockIdx.x * 32 + tl;
  const float4* xr4 = (const float4*)(x1 + (long long)tok * kH);
  float acc = 0.f;
  for (int k4 = 0; k4 < kH / 4; k4++) {
    const float4 xv = xr4[k4];
    const int k = k4 * 4;
    acc += xv.x * gw[(k + 0) * kE + e] + xv.y * gw[(k + 1) * kE + e]
         + xv.z * gw[(k + 2) * kE + e] + xv.w * gw[(k + 3) * kE + e];
  }
  lg[tl][e] = acc;
  __syncthreads();
  if (e == 0) {
    float l[8];
#pragma unroll
    for (int i = 0; i < 8; i++) l[i] = lg[tl][i];
    int i1 = 0; float b1v = l[0];
#pragma unroll
    for (int i = 1; i < 8; i++) if (l[i] > b1v) { b1v = l[i]; i1 = i; }
    float den = 0.f; float ex[8];
#pragma unroll
    for (int i = 0; i < 8; i++) { ex[i] = __expf(l[i] - b1v); den += ex[i]; }
    int i2 = 0; float b2v = -3.0e38f;
#pragma unroll
    for (int i = 0; i < 8; i++) if (i != i1 && l[i] > b2v) { b2v = l[i]; i2 = i; }
    idx1[tok] = i1; idx2[tok] = i2;
    g1[tok] = ex[i1] / den;
    g2[tok] = ex[i2] / den;
  }
}

__global__ void init_slots_kernel(int* __restrict__ tok4slot) {
  const int i = blockIdx.x * 256 + threadIdx.x;
  if (i < kE * kCAP) tok4slot[i] = -1;
}

// ---------------------------------------------------------------------------
// sequential per-expert cumsum (matches reference cumsum/capacity semantics,
// including loc2 offset by total first-choice count), then per-token weights.
// single block of 256 threads.
// ---------------------------------------------------------------------------
__launch_bounds__(256)
__global__ void moe_scan_kernel(const int* __restrict__ idx1, const int* __restrict__ idx2,
                                const float* __restrict__ g1, const float* __restrict__ g2,
                                int* __restrict__ pos1, int* __restrict__ pos2,
                                int* __restrict__ slot1, int* __restrict__ slot2,
                                float* __restrict__ wt1, float* __restrict__ wt2,
                                int* __restrict__ tok4slot) {
  __shared__ int tot1[kE];
  const int t = threadIdx.x;
  if (t < kE) {
    int c = 0;
    for (int s = 0; s < kBS; s++) if (idx1[s] == t) c++;
    tot1[t] = c;
  }
  __syncthreads();
  if (t < kE) {
    int c1 = 0, c2 = tot1[t];
    for (int s = 0; s < kBS; s++) {
      if (idx1[s] == t) pos1[s] = c1++;
      if (idx2[s] == t) pos2[s] = c2++;
    }
  }
  __syncthreads();
  for (int s = t; s < kBS; s += 256) {
    const int p1 = pos1[s], p2 = pos2[s];
    const bool k1 = p1 < kCAP, k2 = p2 < kCAP;
    const float a  = k1 ? g1[s] : 0.f;
    const float bb = k2 ? g2[s] : 0.f;
    const float den = fmaxf(a + bb, 1.1920929e-7f);
    wt1[s] = a / den;  wt2[s] = bb / den;
    slot1[s] = k1 ? p1 : -1;
    slot2[s] = k2 ? p2 : -1;
    if (k1) tok4slot[idx1[s] * kCAP + p1] = s;
    if (k2) tok4slot[idx2[s] * kCAP + p2] = s;
  }
}

// gather tokens into dispatch buffer [E, CAP, H] (bf16); zero empty slots
__global__ void dispatch_kernel(const bf16* __restrict__ x1b, const int* __restrict__ tok4slot,
                                bf16* __restrict__ disp) {
  const int slot = blockIdx.x;
  const int tok = tok4slot[slot];
  bf16* drow = disp + (long long)slot * kH;
  if (tok >= 0) {
    const bf16* srow = x1b + (long long)tok * kH;
    for (int i = threadIdx.x; i < kH; i += 256) drow[i] = srow[i];
  } else {
    const bf16 z = __builtin_bit_cast(bf16, (unsigned short)0);
    for (int i = threadIdx.x; i < kH; i += 256) drow[i] = z;
  }
}

// weighted combine of expert outputs back to token order
__global__ void combine_kernel(const float* __restrict__ expo,
                               const int* __restrict__ e1, const int* __restrict__ e2,
                               const int* __restrict__ s1, const int* __restrict__ s2,
                               const float* __restrict__ wt1, const float* __restrict__ wt2,
                               float* __restrict__ moe) {
  const int tok = blockIdx.x;
  const int a = s1[tok], b2 = s2[tok];
  const float wa = wt1[tok], wb = wt2[tok];
  const long long ba = ((long long)e1[tok] * kCAP + a)  * kH;
  const long long bb = ((long long)e2[tok] * kCAP + b2) * kH;
  for (int i = threadIdx.x; i < kH; i += 256) {
    float v = 0.f;
    if (a  >= 0) v += wa * expo[ba + i];
    if (b2 >= 0) v += wb * expo[bb + i];
    moe[(long long)tok * kH + i] = v;
  }
}

// ---------------------------------------------------------------------------
extern "C" void kernel_launch(void* const* d_in, const int* in_sizes, int n_in,
                              void* d_out, int out_size, void* d_ws, size_t ws_size,
                              hipStream_t stream) {
  (void)in_sizes; (void)n_in; (void)out_size; (void)ws_size;

  const float* x      = (const float*)d_in[0];
  const float* in_w   = (const float*)d_in[1];
  const float* in_b   = (const float*)d_in[2];
  const float* out_w  = (const float*)d_in[3];
  const float* out_b  = (const float*)d_in[4];
  const float* ln1w   = (const float*)d_in[5];
  const float* ln1b   = (const float*)d_in[6];
  const float* ln2w   = (const float*)d_in[7];
  const float* ln2b   = (const float*)d_in[8];
  const float* gate_w = (const float*)d_in[9];
  const float* w1     = (const float*)d_in[10];
  const float* b1     = (const float*)d_in[11];
  const float* w2     = (const float*)d_in[12];
  const float* b2     = (const float*)d_in[13];
  float* out = (float*)d_out;

  char* ws = (char*)d_ws;
  size_t off = 0;
  auto alloc = [&](size_t bytes) -> void* {
    void* p = ws + off;
    off += (bytes + 255) & ~(size_t)255;
    return p;
  };

  bf16*  xb     = (bf16*) alloc((size_t)kBS * kH * 2);
  bf16*  wqkvb  = (bf16*) alloc((size_t)kH3 * kH * 2);
  bf16*  woutb  = (bf16*) alloc((size_t)kH * kH * 2);
  bf16*  w1b    = (bf16*) alloc((size_t)kE * kH * kF * 2);
  bf16*  w2b    = (bf16*) alloc((size_t)kE * kF * kH * 2);
  bf16*  qkvb   = (bf16*) alloc((size_t)kBS * kH3 * 2);
  bf16*  attnb  = (bf16*) alloc((size_t)kBS * kH * 2);
  float* attnpf = (float*)alloc((size_t)kBS * kH * 4);
  float* x1f    = (float*)alloc((size_t)kBS * kH * 4);
  bf16*  x1b    = (bf16*) alloc((size_t)kBS * kH * 2);
  float* moef   = (float*)alloc((size_t)kBS * kH * 4);
  bf16*  dispin = (bf16*) alloc((size_t)kE * kCAP * kH * 2);
  bf16*  hb     = (bf16*) alloc((size_t)kE * kCAP * kF * 2);
  float* expo   = (float*)alloc((size_t)kE * kCAP * kH * 4);
  int*   idx1   = (int*)  alloc((size_t)kBS * 4);
  int*   idx2   = (int*)  alloc((size_t)kBS * 4);
  int*   pos1   = (int*)  alloc((size_t)kBS * 4);
  int*   pos2   = (int*)  alloc((size_t)kBS * 4);
  int*   slot1  = (int*)  alloc((size_t)kBS * 4);
  int*   slot2  = (int*)  alloc((size_t)kBS * 4);
  float* g1     = (float*)alloc((size_t)kBS * 4);
  float* g2     = (float*)alloc((size_t)kBS * 4);
  float* wt1    = (float*)alloc((size_t)kBS * 4);
  float* wt2    = (float*)alloc((size_t)kBS * 4);
  int* tok4slot = (int*)  alloc((size_t)kE * kCAP * 4);

  // --- precision conversion (n/8 vector steps) ---
  cvt_kernel<<<1024, 256, 0, stream>>>(x,     xb,    (long long)kBS * kH / 8);
  cvt_kernel<<<1024, 256, 0, stream>>>(in_w,  wqkvb, (long long)kH3 * kH / 8);
  cvt_kernel<<<512,  256, 0, stream>>>(out_w, woutb, (long long)kH * kH / 8);
  cvt_kernel<<<2048, 256, 0, stream>>>(w1,    w1b,   (long long)kE * kH * kF / 8);
  cvt_kernel<<<2048, 256, 0, stream>>>(w2,    w2b,   (long long)kE * kF * kH / 8);

  // --- qkv = x @ in_proj_w^T + b ---
  gemm_wmma_kernel<true, 0, true><<<dim3(kH3 / 128, kBS / 128, 1), 256, 0, stream>>>(
      xb, wqkvb, in_b, nullptr, qkvb, kH3, kH, 0, 0, 0, 0);

  // --- flash attention ---
  attn_flash_kernel<<<dim3(kS / 128, kNH, kB), 256, 0, stream>>>(qkvb, attnb);

  // --- out_proj ---
  gemm_wmma_kernel<true, 0, false><<<dim3(kH / 128, kBS / 128, 1), 256, 0, stream>>>(
      attnb, woutb, out_b, attnpf, nullptr, kH, kH, 0, 0, 0, 0);

  // --- residual + LN1 ---
  add_ln_kernel<<<kBS, 256, 0, stream>>>(x, attnpf, ln1w, ln1b, x1f, x1b);

  // --- MoE gating / dispatch ---
  gate_kernel<<<kBS / 32, 256, 0, stream>>>(x1f, gate_w, idx1, idx2, g1, g2);
  init_slots_kernel<<<(kE * kCAP + 255) / 256, 256, 0, stream>>>(tok4slot);
  moe_scan_kernel<<<1, 256, 0, stream>>>(idx1, idx2, g1, g2, pos1, pos2,
                                         slot1, slot2, wt1, wt2, tok4slot);
  dispatch_kernel<<<kE * kCAP, 256, 0, stream>>>(x1b, tok4slot, dispin);

  // --- expert FFN: h = gelu(disp @ w1 + b1) ---
  gemm_wmma_kernel<false, 1, true><<<dim3(kF / 128, kCAP / 128, kE), 256, 0, stream>>>(
      dispin, w1b, b1, nullptr, hb, kF, kH,
      (long long)kCAP * kH, (long long)kH * kF, (long long)kF, (long long)kCAP * kF);

  // --- expert FFN: exp_out = h @ w2 + b2 ---
  gemm_wmma_kernel<false, 0, false><<<dim3(kH / 128, kCAP / 128, kE), 256, 0, stream>>>(
      hb, w2b, b2, expo, nullptr, kH, kF,
      (long long)kCAP * kF, (long long)kF * kH, (long long)kH, (long long)kCAP * kH);

  // --- combine + residual + LN2 ---
  combine_kernel<<<kBS, 256, 0, stream>>>(expo, idx1, idx2, slot1, slot2, wt1, wt2, moef);
  add_ln_kernel<<<kBS, 256, 0, stream>>>(x1f, moef, ln2w, ln2b, out, nullptr);
}
